// AttentionHelper_69398081569424
// MI455X (gfx1250) — compile-verified, matrix operand order checked by reading, not run on hardware
//
#include <hip/hip_runtime.h>
#include <math.h>

typedef __attribute__((ext_vector_type(16))) _Float16 v16h;
typedef __attribute__((ext_vector_type(8)))  float    v8f;

#define B_    16
#define C_    256
#define L_    2048
#define QBLK  128          // queries per workgroup (8 waves x 16)
#define KT    32           // keys per tile
#define QROW  264          // padded halves per Q/K LDS row (528B = 132 dwords -> conflict-free frags)
#define VROW  40           // padded halves per V LDS row   (80B  = 20 dwords  -> conflict-free frags)
#define SCALE 0.0625f      // 1/sqrt(256)

// LDS partition (bytes), all 16B aligned
#define QS_OFF   0
#define KS_OFF   (QBLK * QROW * 2)                 // 67584
#define VS_OFF   (KS_OFF + KT * QROW * 2)          // 84480
#define LM_OFF   (VS_OFF + C_ * VROW * 2)          // 104960
#define PM_OFF   (LM_OFF + KT * 4)                 // 105088
#define SMEM_SZ  (PM_OFF + KT * 4)                 // 105216

// Load a 16-bit A/B WMMA fragment (16x32, K-chunk layout) from a padded row-major
// f16 LDS buffer. Per ISA: lanes 0-15 hold K={0..7,16..23}(+8 for lanes 16-31) pairs.
// p points at row + 8*half within the 32-wide K chunk.
__device__ __forceinline__ v16h load_frag(const _Float16* p) {
    union { int4 i[2]; v16h h; } u;
    u.i[0] = *(const int4*)(p);
    u.i[1] = *(const int4*)(p + 16);
    return u.h;
}

__device__ __forceinline__ float xor16(float x) {
    // swap lane halves (SWAPX16): offset = {xor=0x10, or=0, and=0x1f}
    return __int_as_float(__builtin_amdgcn_ds_swizzle(__float_as_int(x), 0x401F));
}

extern "C" __global__ __launch_bounds__(256)
void flash_attn_gfx1250(const float* __restrict__ Qg, const float* __restrict__ Kg,
                        const float* __restrict__ Vg, const float* __restrict__ Mg,
                        float* __restrict__ Og)
{
    extern __shared__ char smem[];
    _Float16* Qs    = (_Float16*)(smem + QS_OFF);   // [QBLK][QROW]  (l-major, c fast)
    _Float16* Ks    = (_Float16*)(smem + KS_OFF);   // [KT][QROW]    (m-major, c fast)
    _Float16* Vs    = (_Float16*)(smem + VS_OFF);   // [C_][VROW]    (c-major, m fast)
    float*    lmask = (float*)(smem + LM_OFF);      // log(mask+1e-9) per key of tile
    float*    pmask = (float*)(smem + PM_OFF);      // mask per key of tile

    const int t    = threadIdx.x;
    const int lane = t & 31;
    const int wave = t >> 5;
    const int half = lane >> 4;
    const int lr   = lane & 15;
    const int b    = blockIdx.y;
    const int l0   = blockIdx.x * QBLK;

    const float* Qb = Qg + (size_t)b * C_ * L_;
    const float* Kb = Kg + (size_t)b * C_ * L_;
    const float* Vb = Vg + (size_t)b * C_ * L_;
    const float* Mb = Mg + (size_t)b * L_;

    const int cr = t >> 3;         // row-within-pass 0..31
    const int cq = (t & 7) * 4;    // 4-float column group

    // ---- stage Q block, transposed to [l][c] in f16 (coalesced 128B rows) ----
    #pragma unroll
    for (int pass = 0; pass < 8; ++pass) {
        const int c = pass * 32 + cr;
        #pragma unroll
        for (int seg = 0; seg < 4; ++seg) {
            const int col = seg * 32 + cq;
            float4 q = *(const float4*)(Qb + (size_t)c * L_ + l0 + col);
            Qs[(col + 0) * QROW + c] = (_Float16)q.x;
            Qs[(col + 1) * QROW + c] = (_Float16)q.y;
            Qs[(col + 2) * QROW + c] = (_Float16)q.z;
            Qs[(col + 3) * QROW + c] = (_Float16)q.w;
        }
    }
    __syncthreads();

    // ---- persistent Q B-fragments: lane = query column l, K = channel c ----
    v16h qf[8];
    {
        const _Float16* qrow = Qs + (wave * 16 + lr) * QROW + 8 * half;
        #pragma unroll
        for (int cc = 0; cc < 8; ++cc) qf[cc] = load_frag(qrow + cc * 32);
    }

    v8f acc[16];                                // O^T accumulators: 16 c-chunks x (16c x 16l)
    #pragma unroll
    for (int i = 0; i < 16; ++i) { v8f z = {0,0,0,0,0,0,0,0}; acc[i] = z; }
    float run_m = -3.0e38f, run_s = 0.0f;

    for (int kt = 0; kt < L_ / KT; ++kt) {
        const int m0 = kt * KT;
        __syncthreads();   // previous tile fully consumed

        // ---- stage K tile (transposed) and V tile, f32->f16, coalesced ----
        #pragma unroll
        for (int pass = 0; pass < 8; ++pass) {
            const int c = pass * 32 + cr;
            float4 kv = *(const float4*)(Kb + (size_t)c * L_ + m0 + cq);
            Ks[(cq + 0) * QROW + c] = (_Float16)kv.x;
            Ks[(cq + 1) * QROW + c] = (_Float16)kv.y;
            Ks[(cq + 2) * QROW + c] = (_Float16)kv.z;
            Ks[(cq + 3) * QROW + c] = (_Float16)kv.w;
            float4 vv = *(const float4*)(Vb + (size_t)c * L_ + m0 + cq);
            union { _Float16 h[4]; int2 q; } vh;
            vh.h[0] = (_Float16)vv.x; vh.h[1] = (_Float16)vv.y;
            vh.h[2] = (_Float16)vv.z; vh.h[3] = (_Float16)vv.w;
            *(int2*)(Vs + (size_t)c * VROW + cq) = vh.q;
            // hint next tile into cache (speculative; OOB silently dropped)
            __builtin_prefetch(Kb + (size_t)c * L_ + m0 + KT + cq, 0, 0);
            __builtin_prefetch(Vb + (size_t)c * L_ + m0 + KT + cq, 0, 0);
        }
        if (t < KT) {
            float mv = Mb[m0 + t];
            lmask[t] = __logf(mv + 1e-9f);
            pmask[t] = mv;
        }
        __syncthreads();

        // ---- energy (transposed): S^T = K_tile x Q,  D[m,l] ----
        v8f s0 = {0,0,0,0,0,0,0,0}, s1 = {0,0,0,0,0,0,0,0};
        {
            const _Float16* k0 = Ks + lr * QROW + 8 * half;          // keys m0+0..15
            const _Float16* k1 = Ks + (16 + lr) * QROW + 8 * half;   // keys m0+16..31
            #pragma unroll
            for (int cc = 0; cc < 8; ++cc) {
                v16h a0 = load_frag(k0 + cc * 32);
                v16h a1 = load_frag(k1 + cc * 32);
                s0 = __builtin_amdgcn_wmma_f32_16x16x32_f16(false, a0, false, qf[cc], (short)0, s0, false, false);
                s1 = __builtin_amdgcn_wmma_f32_16x16x32_f16(false, a1, false, qf[cc], (short)0, s1, false, false);
            }
        }

        // ---- online softmax over this tile's 32 keys (per query l = lane) ----
        // lane holds m = v + 8*half (s0) and m = 16 + v + 8*half (s1)
        float e0[8], e1[8];
        float mx = -3.0e38f;
        #pragma unroll
        for (int v = 0; v < 8; ++v) {
            e0[v] = s0[v] * SCALE + lmask[v + 8 * half];
            e1[v] = s1[v] * SCALE + lmask[16 + v + 8 * half];
            mx = fmaxf(mx, fmaxf(e0[v], e1[v]));
        }
        mx = fmaxf(mx, xor16(mx));
        const float nm   = fmaxf(run_m, mx);
        const float corr = __expf(run_m - nm);
        float p0[8], p1[8], ts = 0.0f;
        #pragma unroll
        for (int v = 0; v < 8; ++v) {
            p0[v] = __expf(e0[v] - nm);
            p1[v] = __expf(e1[v] - nm);
            ts += p0[v] + p1[v];
        }
        ts += xor16(ts);
        run_s = run_s * corr + ts;
        run_m = nm;

        // ---- pack P * post-softmax mask directly as WMMA B-fragment ----
        union { v16h h; int4 i[2]; } bp;
        #pragma unroll
        for (int v = 0; v < 8; ++v) {
            bp.h[v]     = (_Float16)(p0[v] * pmask[v + 8 * half]);
            bp.h[8 + v] = (_Float16)(p1[v] * pmask[16 + v + 8 * half]);
        }

        // ---- O^T[c,l] = V_tile(16c x 32m) x P(32m x 16l) + corr*O^T ----
        #pragma unroll
        for (int cc2 = 0; cc2 < 16; ++cc2) {
            v8f cacc = acc[cc2];
            #pragma unroll
            for (int i = 0; i < 8; ++i) cacc[i] *= corr;
            v16h av = load_frag(Vs + (size_t)(cc2 * 16 + lr) * VROW + 8 * half);
            acc[cc2] = __builtin_amdgcn_wmma_f32_16x16x32_f16(false, av, false, bp.h, (short)0, cacc, false, false);
        }
    }

    // ---- write out[b, c, l] = O^T[c,l] / Z[l] ----
    const float inv = 1.0f / run_s;
    float* outp = Og + (size_t)b * C_ * L_ + (l0 + wave * 16 + lr);
    #pragma unroll
    for (int cc2 = 0; cc2 < 16; ++cc2) {
        #pragma unroll
        for (int v = 0; v < 8; ++v) {
            const int c = cc2 * 16 + v + 8 * half;
            outp[(size_t)c * L_] = acc[cc2][v] * inv;
        }
    }
}

extern "C" void kernel_launch(void* const* d_in, const int* in_sizes, int n_in,
                              void* d_out, int out_size, void* d_ws, size_t ws_size,
                              hipStream_t stream) {
    (void)in_sizes; (void)n_in; (void)d_ws; (void)ws_size; (void)out_size;
    const float* Q = (const float*)d_in[0];
    const float* K = (const float*)d_in[1];
    const float* V = (const float*)d_in[2];
    const float* M = (const float*)d_in[3];
    float* O = (float*)d_out;

    dim3 grid(L_ / QBLK, B_);   // 16 x 16 = 256 workgroups
    dim3 block(256);            // 8 wave32
    hipLaunchKernelGGL(flash_attn_gfx1250, grid, block, SMEM_SZ, stream, Q, K, V, M, O);
}
